// CausalSelfAttention_33414845563556
// MI455X (gfx1250) — matrix operand order repaired and checked by reading, not compiled, not run on hardware
//
#include <hip/hip_runtime.h>

// ---------------------------------------------------------------------------
// CDNA5 (gfx1250) causal self-attention: bf16 WMMA GEMMs + fp32 flash softmax
//   B=4, T=2048, C=1024, H=16, D=64
//   - one-time fp32 -> bf16 conversion of x / W_attn / W_proj
//   - GEMMs: v_wmma_f32_16x16x32_bf16, TDM double-buffered tile streaming
//   - attention: TDM tile staging + fp32 online softmax
// ---------------------------------------------------------------------------

typedef __attribute__((ext_vector_type(16))) __bf16 v16bf;
typedef __attribute__((ext_vector_type(8)))  float  v8f;
typedef __attribute__((ext_vector_type(4)))  unsigned int tdm4;
typedef __attribute__((ext_vector_type(8)))  unsigned int tdm8;

#define WMMA_BF16(A, B, C) \
  __builtin_amdgcn_wmma_f32_16x16x32_bf16(false, (A), false, (B), (short)0, (C), false, false)

// float -> bf16 bits via hardware cvt (RTE)
__device__ __forceinline__ unsigned short f2bf_bits(float f) {
  return __builtin_bit_cast(unsigned short, (__bf16)f);
}
__device__ __forceinline__ __bf16 bits2bf(unsigned short b) {
  return __builtin_bit_cast(__bf16, b);
}

// ---------------------------------------------------------------------------
// Tensor Data Mover: DMA a rows x cols bf16 tile (row stride = stride_elems)
// from global memory into LDS at lds_off_bytes. D# built per CDNA5 ISA §8.
// 2D tile -> descriptor groups 2/3 omitted (VADDR2/3 = NULL form).
// ---------------------------------------------------------------------------
__device__ __forceinline__ void tdm_load_tile_bf16(const unsigned short* gsrc,
                                                   unsigned lds_off_bytes,
                                                   unsigned rows, unsigned cols,
                                                   unsigned stride_elems) {
  const unsigned long long ga = (unsigned long long)(size_t)gsrc;
  tdm4 g0;
  g0[0] = 1u;                                   // count=1, load, user mode
  g0[1] = lds_off_bytes;                        // lds_addr
  g0[2] = (unsigned)ga;                         // global_addr[31:0]
  g0[3] = ((unsigned)(ga >> 32) & 0x01FFFFFFu)  // global_addr[56:32]
          | (2u << 30);                         // type=2 ("image")
  tdm8 g1;
  g1[0] = 1u << 16;                             // data_size=1 (2 bytes)
  g1[1] = (cols & 0xFFFFu) << 16;               // tensor_dim0[15:0]
  g1[2] = (cols >> 16) | ((rows & 0xFFFFu) << 16);  // dim0 hi | tensor_dim1 lo
  g1[3] = (rows >> 16) | ((cols & 0xFFFFu) << 16);  // dim1 hi | tile_dim0
  g1[4] = rows & 0xFFFFu;                       // tile_dim1 (tile_dim2 = 0)
  g1[5] = stride_elems;                         // tensor_dim0_stride[31:0]
  g1[6] = 0u;                                   // stride hi | dim1_stride lo
  g1[7] = 0u;                                   // dim1_stride hi
  asm volatile("tensor_load_to_lds %0, %1" :: "s"(g0), "s"(g1) : "memory");
}

// --- A fragment (16x32, bf16), row-major LDS src with given stride ---------
__device__ __forceinline__ v16bf load_a_frag(const unsigned short* lds,
                                             int row_base, int stride, int kk) {
  const int lane = threadIdx.x & 31;
  const int lh   = lane >> 4;
  const int m    = lane & 15;
  v16bf a;
#pragma unroll
  for (int i = 0; i < 8; ++i) {
    const int k = 16 * lh + 2 * i;
    const unsigned short* p = lds + (size_t)(row_base + m) * stride + kk + k;
    a[2 * i]     = bits2bf(p[0]);
    a[2 * i + 1] = bits2bf(p[1]);
  }
  return a;
}

// --- B fragment (32x16, bf16) from row-major [K][N] LDS --------------------
__device__ __forceinline__ v16bf load_b_frag(const unsigned short* lds,
                                             int col_base, int stride, int kk) {
  const int lane = threadIdx.x & 31;
  const int lh   = lane >> 4;
  const int n    = lane & 15;
  v16bf b;
#pragma unroll
  for (int v = 0; v < 8; ++v) {
    const int k = 16 * lh + 2 * v;
    b[2 * v]     = bits2bf(lds[(size_t)(kk + k)     * stride + col_base + n]);
    b[2 * v + 1] = bits2bf(lds[(size_t)(kk + k + 1) * stride + col_base + n]);
  }
  return b;
}

// --- B fragment (32x16) where memory holds B transposed: src[n][k] ---------
__device__ __forceinline__ v16bf load_b_frag_T(const unsigned short* lds,
                                               int col_base, int stride, int kk) {
  const int lane = threadIdx.x & 31;
  const int lh   = lane >> 4;
  const int n    = lane & 15;
  v16bf b;
#pragma unroll
  for (int v = 0; v < 8; ++v) {
    const int k = 16 * lh + 2 * v;
    const unsigned short* p = lds + (size_t)(col_base + n) * stride + kk + k;
    b[2 * v]     = bits2bf(p[0]);
    b[2 * v + 1] = bits2bf(p[1]);
  }
  return b;
}

// ---------------------------------------------------------------------------
// fp32 -> bf16 bulk conversion (float4 in, ushort4 out), n4 = n/4
// ---------------------------------------------------------------------------
__global__ void __launch_bounds__(256)
cvt_f32_bf16_kernel(const float* __restrict__ src, unsigned short* __restrict__ dst,
                    int n4) {
  const int i = blockIdx.x * 256 + threadIdx.x;
  if (i < n4) {
    const float4 f = reinterpret_cast<const float4*>(src)[i];
    ushort4 o;
    o.x = f2bf_bits(f.x);
    o.y = f2bf_bits(f.y);
    o.z = f2bf_bits(f.z);
    o.w = f2bf_bits(f.w);
    reinterpret_cast<ushort4*>(dst)[i] = o;
  }
}

// ---------------------------------------------------------------------------
// GEMM: out[M,N] = A[M,K] * W[K,N] + bias[N]   (A, W already bf16)
// Block: 256 thr = 8 waves; tile 64(M) x 128(N), K-step 32; wave = 32x32.
// TDM double-buffered: wave 0 DMAs next K-tile while all waves run WMMA.
// ---------------------------------------------------------------------------
template <bool BF16OUT>
__global__ void __launch_bounds__(256)
gemm_bias_kernel(const unsigned short* __restrict__ A,
                 const unsigned short* __restrict__ W,
                 const float* __restrict__ bias, void* __restrict__ outp,
                 int M, int N, int K) {
  __shared__ unsigned short sA[2][64 * 32];
  __shared__ unsigned short sB[2][32 * 128];

  const int tid    = threadIdx.x;
  const int lane   = tid & 31;
  const int wid    = tid >> 5;
  const int lh     = lane >> 4;
  const int l15    = lane & 15;
  const int wave_m = wid & 1;        // 0..1 -> 32-row stripe
  const int wave_n = wid >> 1;       // 0..3 -> 32-col stripe
  const int m0 = blockIdx.y * 64;
  const int n0 = blockIdx.x * 128;

  v8f acc00 = {}, acc01 = {}, acc10 = {}, acc11 = {};

  const int nkt = K >> 5;  // number of 32-deep K tiles

  // prologue: DMA tile 0 into buffer 0
  if (wid == 0) {
    tdm_load_tile_bf16(A + (size_t)m0 * K, (unsigned)(size_t)&sA[0][0], 64, 32, K);
    tdm_load_tile_bf16(W + n0,             (unsigned)(size_t)&sB[0][0], 32, 128, N);
  }

  for (int it = 0; it < nkt; ++it) {
    const int cur = it & 1;
    __syncthreads();  // all readers done with buffer cur^1 (prev iteration)
    if (wid == 0) {
      if (it + 1 < nkt) {
        const int kb2 = (it + 1) << 5;
        tdm_load_tile_bf16(A + (size_t)m0 * K + kb2,
                           (unsigned)(size_t)&sA[cur ^ 1][0], 64, 32, K);
        tdm_load_tile_bf16(W + (size_t)kb2 * N + n0,
                           (unsigned)(size_t)&sB[cur ^ 1][0], 32, 128, N);
        // TENSORcnt completes in order: <=2 outstanding => current tile landed
        __builtin_amdgcn_s_wait_tensorcnt(2);
      } else {
        __builtin_amdgcn_s_wait_tensorcnt(0);
      }
    }
    __syncthreads();  // publish buffer cur

    const unsigned short* cA = sA[cur];
    const unsigned short* cB = sB[cur];
    const v16bf a0 = load_a_frag(cA, wave_m * 32,      32, 0);
    const v16bf a1 = load_a_frag(cA, wave_m * 32 + 16, 32, 0);
    const v16bf b0 = load_b_frag(cB, wave_n * 32,      128, 0);
    const v16bf b1 = load_b_frag(cB, wave_n * 32 + 16, 128, 0);
    acc00 = WMMA_BF16(a0, b0, acc00);
    acc01 = WMMA_BF16(a0, b1, acc01);
    acc10 = WMMA_BF16(a1, b0, acc10);
    acc11 = WMMA_BF16(a1, b1, acc11);
  }

  // epilogue: C/D layout row = r + 8*lh, col = l15
#pragma unroll
  for (int sm = 0; sm < 2; ++sm) {
#pragma unroll
    for (int sn = 0; sn < 2; ++sn) {
      const v8f& acc = (sm == 0) ? (sn == 0 ? acc00 : acc01)
                                 : (sn == 0 ? acc10 : acc11);
#pragma unroll
      for (int r = 0; r < 8; ++r) {
        const int row = m0 + wave_m * 32 + sm * 16 + r + 8 * lh;
        const int col = n0 + wave_n * 32 + sn * 16 + l15;
        const float v = acc[r] + bias[col];
        if constexpr (BF16OUT)
          ((unsigned short*)outp)[(size_t)row * N + col] = f2bf_bits(v);
        else
          ((float*)outp)[(size_t)row * N + col] = v;
      }
    }
  }
}

// ---------------------------------------------------------------------------
// Flash attention: block = (q-tile 64 rows, head, batch), 128 thr = 4 waves.
// Q/K/V tiles DMA'd into LDS by the Tensor Data Mover.
// ---------------------------------------------------------------------------
__global__ void __launch_bounds__(128)
attn_kernel(const unsigned short* __restrict__ qkv,  // [B*T, 3072] bf16
            unsigned short* __restrict__ ctx) {      // [B*T, 1024] bf16
  __shared__ unsigned short sQ[64 * 64];
  __shared__ unsigned short sK[64 * 64];
  __shared__ unsigned short sV[64 * 64];
  __shared__ unsigned short sP[4 * 16 * 64];  // wave-private P stripes

  const int tid  = threadIdx.x;
  const int lane = tid & 31;
  const int wid  = tid >> 5;      // 0..3 -> q rows [16*wid, 16*wid+16)
  const int lh   = lane >> 4;
  const int l15  = lane & 15;

  const int qt = blockIdx.x;      // 0..31
  const int h  = blockIdx.y;      // 0..15
  const int b  = blockIdx.z;      // 0..3
  const int q0 = qt * 64;
  const size_t bT = (size_t)b * 2048;
  const int qoff = h * 64, koff = 1024 + h * 64, voff = 2048 + h * 64;

  const unsigned ldsQ = (unsigned)(size_t)&sQ[0];
  const unsigned ldsK = (unsigned)(size_t)&sK[0];
  const unsigned ldsV = (unsigned)(size_t)&sV[0];

  // async-stage Q tile via TDM (waited together with first K/V tiles)
  if (wid == 0)
    tdm_load_tile_bf16(qkv + (bT + q0) * 3072 + qoff, ldsQ, 64, 64, 3072);

  v8f o0 = {}, o1 = {}, o2 = {}, o3 = {};
  float mrow[8], lrow[8];
#pragma unroll
  for (int r = 0; r < 8; ++r) { mrow[r] = -__builtin_inff(); lrow[r] = 0.f; }

  for (int kt = 0; kt <= qt; ++kt) {
    const int k0 = kt * 64;
    __syncthreads();  // previous iteration's readers done with sK/sV
    if (wid == 0) {
      tdm_load_tile_bf16(qkv + (bT + k0) * 3072 + koff, ldsK, 64, 64, 3072);
      tdm_load_tile_bf16(qkv + (bT + k0) * 3072 + voff, ldsV, 64, 64, 3072);
      __builtin_amdgcn_s_wait_tensorcnt(0);  // TDM -> LDS complete
    }
    __syncthreads();  // publish tiles to all waves

    // ---- S = Q * K^T  (16 q-rows x 64 keys per wave, D=64 -> 2 K-steps) ----
    v8f s0 = {}, s1 = {}, s2 = {}, s3 = {};
#pragma unroll
    for (int kk = 0; kk < 64; kk += 32) {
      const v16bf aq  = load_a_frag(sQ, 16 * wid, 64, kk);
      const v16bf bk0 = load_b_frag_T(sK,  0, 64, kk);
      const v16bf bk1 = load_b_frag_T(sK, 16, 64, kk);
      const v16bf bk2 = load_b_frag_T(sK, 32, 64, kk);
      const v16bf bk3 = load_b_frag_T(sK, 48, 64, kk);
      s0 = WMMA_BF16(aq, bk0, s0);
      s1 = WMMA_BF16(aq, bk1, s1);
      s2 = WMMA_BF16(aq, bk2, s2);
      s3 = WMMA_BF16(aq, bk3, s3);
    }

    // ---- scale + causal mask + online softmax (fp32) ----
    const float scale = 0.125f;  // 1/sqrt(64)
    const bool diag = (kt == qt);
#pragma unroll
    for (int r = 0; r < 8; ++r) {
      const int m = 16 * wid + r + 8 * lh;  // local q row within 64-block
      float v0 = s0[r] * scale, v1 = s1[r] * scale,
            v2 = s2[r] * scale, v3 = s3[r] * scale;
      if (diag) {
        if ( 0 + l15 > m) v0 = -__builtin_inff();
        if (16 + l15 > m) v1 = -__builtin_inff();
        if (32 + l15 > m) v2 = -__builtin_inff();
        if (48 + l15 > m) v3 = -__builtin_inff();
      }
      float mx = fmaxf(fmaxf(v0, v1), fmaxf(v2, v3));
      mx = fmaxf(mx, __shfl_xor(mx, 1));
      mx = fmaxf(mx, __shfl_xor(mx, 2));
      mx = fmaxf(mx, __shfl_xor(mx, 4));
      mx = fmaxf(mx, __shfl_xor(mx, 8));
      const float mnew  = fmaxf(mrow[r], mx);
      const float alpha = __expf(mrow[r] - mnew);
      const float p0 = __expf(v0 - mnew), p1 = __expf(v1 - mnew);
      const float p2 = __expf(v2 - mnew), p3 = __expf(v3 - mnew);
      float ps = p0 + p1 + p2 + p3;
      ps += __shfl_xor(ps, 1);
      ps += __shfl_xor(ps, 2);
      ps += __shfl_xor(ps, 4);
      ps += __shfl_xor(ps, 8);
      lrow[r] = lrow[r] * alpha + ps;
      mrow[r] = mnew;
      o0[r] *= alpha; o1[r] *= alpha; o2[r] *= alpha; o3[r] *= alpha;
      // spill P (bf16) into this wave's private LDS stripe for A-frag reload
      unsigned short* pl = sP + wid * 1024 + (r + 8 * lh) * 64;
      pl[ 0 + l15] = f2bf_bits(p0);
      pl[16 + l15] = f2bf_bits(p1);
      pl[32 + l15] = f2bf_bits(p2);
      pl[48 + l15] = f2bf_bits(p3);
    }
    // intra-wave LDS handoff: wait for our DS stores before re-reading
    asm volatile("s_wait_dscnt 0x0" ::: "memory");
    __builtin_amdgcn_wave_barrier();

    // ---- O += P * V  (16 x 64 += (16x64)*(64x64), 2 K-steps) ----
#pragma unroll
    for (int kk = 0; kk < 64; kk += 32) {
      const v16bf ap  = load_a_frag(sP + wid * 1024, 0, 64, kk);
      const v16bf bv0 = load_b_frag(sV,  0, 64, kk);
      const v16bf bv1 = load_b_frag(sV, 16, 64, kk);
      const v16bf bv2 = load_b_frag(sV, 32, 64, kk);
      const v16bf bv3 = load_b_frag(sV, 48, 64, kk);
      o0 = WMMA_BF16(ap, bv0, o0);
      o1 = WMMA_BF16(ap, bv1, o1);
      o2 = WMMA_BF16(ap, bv2, o2);
      o3 = WMMA_BF16(ap, bv3, o3);
    }
  }

  // ---- normalize and store context [B*T, H*D] as bf16 ----
#pragma unroll
  for (int r = 0; r < 8; ++r) {
    const float rinv = 1.f / lrow[r];
    const int qrow = q0 + 16 * wid + r + 8 * lh;
    const size_t base = (bT + qrow) * 1024 + h * 64;
    ctx[base +  0 + l15] = f2bf_bits(o0[r] * rinv);
    ctx[base + 16 + l15] = f2bf_bits(o1[r] * rinv);
    ctx[base + 32 + l15] = f2bf_bits(o2[r] * rinv);
    ctx[base + 48 + l15] = f2bf_bits(o3[r] * rinv);
  }
}

// ---------------------------------------------------------------------------
extern "C" void kernel_launch(void* const* d_in, const int* in_sizes, int n_in,
                              void* d_out, int out_size, void* d_ws, size_t ws_size,
                              hipStream_t stream) {
  (void)in_sizes; (void)n_in; (void)out_size; (void)ws_size;
  const float* x      = (const float*)d_in[0];  // [4,2048,1024]
  const float* W_attn = (const float*)d_in[1];  // [1024,3072]
  const float* b_attn = (const float*)d_in[2];  // [3072]
  const float* W_proj = (const float*)d_in[3];  // [1024,1024]
  const float* b_proj = (const float*)d_in[4];  // [1024]
  float* out = (float*)d_out;                   // [4,2048,1024] fp32

  // workspace layout (bf16 elements)
  unsigned short* qkv_ws = (unsigned short*)d_ws;             // 8192*3072
  unsigned short* ctx_ws = qkv_ws + (size_t)8192 * 3072;      // 8192*1024
  unsigned short* x_bf   = ctx_ws + (size_t)8192 * 1024;      // 8192*1024
  unsigned short* Wa_bf  = x_bf   + (size_t)8192 * 1024;      // 1024*3072
  unsigned short* Wp_bf  = Wa_bf  + (size_t)1024 * 3072;      // 1024*1024

  // 0) one-time fp32 -> bf16 conversion of operands
  const int n4_x  = (8192 * 1024) / 4;
  const int n4_wa = (1024 * 3072) / 4;
  const int n4_wp = (1024 * 1024) / 4;
  cvt_f32_bf16_kernel<<<(n4_x  + 255) / 256, 256, 0, stream>>>(x,      x_bf,  n4_x);
  cvt_f32_bf16_kernel<<<(n4_wa + 255) / 256, 256, 0, stream>>>(W_attn, Wa_bf, n4_wa);
  cvt_f32_bf16_kernel<<<(n4_wp + 255) / 256, 256, 0, stream>>>(W_proj, Wp_bf, n4_wp);

  // 1) qkv = x @ W_attn + b_attn  (bf16 out)
  dim3 g1(3072 / 128, 8192 / 64);
  gemm_bias_kernel<true><<<g1, 256, 0, stream>>>(
      x_bf, Wa_bf, b_attn, qkv_ws, 8192, 3072, 1024);

  // 2) flash attention per (q-tile, head, batch)
  dim3 g2(2048 / 64, 16, 4);
  attn_kernel<<<g2, 128, 0, stream>>>(qkv_ws, ctx_ws);

  // 3) out = ctx @ W_proj + b_proj  (fp32 out)
  dim3 g3(1024 / 128, 8192 / 64);
  gemm_bias_kernel<false><<<g3, 256, 0, stream>>>(
      ctx_ws, Wp_bf, b_proj, out, 8192, 1024, 1024);
}
